// DeformLightVideo_81570018885637
// MI455X (gfx1250) — compile-verified, matrix operand order checked by reading, not run on hardware
//
#include <hip/hip_runtime.h>
#include <hip/hip_bf16.h>
#include <math.h>

#define B_ 4
#define Q_ 64
#define C_ 384
#define H_ 6
#define L_ 5
#define P_ 4
#define DFF_ 1024
#define EXTRA_ 128
#define S_ 66836
#define DH_ 64

typedef __attribute__((ext_vector_type(16))) __bf16 v16bf;
typedef __attribute__((ext_vector_type(8)))  float  v8f;

__device__ __forceinline__ __bf16 f2bf(float f) {
    unsigned u = __builtin_bit_cast(unsigned, f);
    u += 0x7FFFu + ((u >> 16) & 1u);          // round-to-nearest-even
    unsigned short h = (unsigned short)(u >> 16);
    return __builtin_bit_cast(__bf16, h);
}

// ---------------------------------------------------------------------------
// Generic GEMM: Y[M,N] = X[M,K] @ W[N,K]^T + bias[N], fp32 in/out,
// bf16 WMMA (v_wmma_f32_16x16x32_bf16) inside. K must be a multiple of 32.
// Block tile 128(M) x 64(N); 256 threads = 8 waves, each wave owns a
// 16(M) x 64(N) strip = 4 accumulator fragments.
// ---------------------------------------------------------------------------
#define BM 128
#define BN 64
#define BK 32

template<bool RELU>
__global__ __launch_bounds__(256) void gemm_bias_wmma(
    const float* __restrict__ X, const float* __restrict__ W,
    const float* __restrict__ bias, float* __restrict__ Y,
    int M, int N, int K)
{
    __shared__ __bf16 As[BM][BK + 2];
    __shared__ __bf16 Bs[BN][BK + 2];

    const int tid   = threadIdx.x;
    const int wave  = tid >> 5;
    const int lane  = tid & 31;
    const int lhalf = lane >> 4;      // 0 or 1
    const int l16   = lane & 15;

    const int n0 = blockIdx.x * BN;   // N varies fastest -> L2 reuse of A tiles
    const int m0 = blockIdx.y * BM;
    const int mbase = m0 + wave * 16;

    v8f acc0 = {}, acc1 = {}, acc2 = {}, acc3 = {};

    for (int k0 = 0; k0 < K; k0 += BK) {
        // ---- stage A tile (convert fp32 -> bf16) ----
        for (int i = tid; i < BM * BK; i += 256) {
            int m  = i >> 5;          // /BK
            int k  = i & 31;
            int gm = m0 + m;
            float v = (gm < M) ? X[(size_t)gm * K + k0 + k] : 0.f;
            As[m][k] = f2bf(v);
            if ((k0 + BK < K) && (gm < M) && (k == 0))
                __builtin_prefetch(&X[(size_t)gm * K + k0 + BK], 0, 0);
        }
        // ---- stage B tile (W is [N][K] row-major) ----
        for (int i = tid; i < BN * BK; i += 256) {
            int n  = i >> 5;
            int k  = i & 31;
            int gn = n0 + n;
            float v = (gn < N) ? W[(size_t)gn * K + k0 + k] : 0.f;
            Bs[n][k] = f2bf(v);
        }
        __syncthreads();

        // ---- build A fragment (ISA 7.12.2 bf16 16x32 layout) ----
        union { v16bf v; __bf16 e[16]; } af;
        #pragma unroll
        for (int j = 0; j < 16; ++j) {
            int kk = ((j >= 8) ? 16 : 0) + (lhalf ? 8 : 0) + (j & 7);
            af.e[j] = As[wave * 16 + l16][kk];
        }
        // ---- 4 B fragments + WMMA ----
        #pragma unroll
        for (int ns = 0; ns < 4; ++ns) {
            union { v16bf v; __bf16 e[16]; } bfr;
            #pragma unroll
            for (int j = 0; j < 16; ++j) {
                int kk = ((j >= 8) ? 16 : 0) + (lhalf ? 8 : 0) + (j & 7);
                bfr.e[j] = Bs[ns * 16 + l16][kk];
            }
            v8f* accp = (ns == 0) ? &acc0 : (ns == 1) ? &acc1
                       : (ns == 2) ? &acc2 : &acc3;
            *accp = __builtin_amdgcn_wmma_f32_16x16x32_bf16(
                false, af.v, false, bfr.v, (short)0, *accp, false, false);
        }
        __syncthreads();
    }

    // ---- store with fused bias (+ReLU) ----
    union { v8f v; float e[8]; } a0, a1, a2, a3;
    a0.v = acc0; a1.v = acc1; a2.v = acc2; a3.v = acc3;
    #pragma unroll
    for (int ns = 0; ns < 4; ++ns) {
        const float* ep = (ns == 0) ? a0.e : (ns == 1) ? a1.e
                         : (ns == 2) ? a2.e : a3.e;
        int col = n0 + ns * 16 + l16;
        if (col < N) {
            float bv = bias[col];
            #pragma unroll
            for (int r = 0; r < 8; ++r) {
                int row = mbase + r + lhalf * 8;
                if (row < M) {
                    float v = ep[r] + bv;
                    if (RELU) v = fmaxf(v, 0.f);
                    Y[(size_t)row * N + col] = v;
                }
            }
        }
    }
}

// ---------------------------------------------------------------------------
// Fused per-(b,h) scaled-dot attention; thread = query, NK keys (64 or 128).
// K then V staged in LDS (reused buffer).
// ---------------------------------------------------------------------------
template<int NK>
__global__ __launch_bounds__(64) void attn_kernel(
    const float* __restrict__ Qb, int qStride,
    const float* __restrict__ Kb, const float* __restrict__ Vb, int kvStride,
    float* __restrict__ out)
{
    __shared__ float KV[NK][DH_ + 1];
    const int b = blockIdx.x / H_;
    const int h = blockIdx.x % H_;
    const int q = threadIdx.x;

    for (int i = threadIdx.x; i < NK * DH_; i += 64) {
        int r = i / DH_, d = i % DH_;
        KV[r][d] = Kb[(size_t)(b * NK + r) * kvStride + h * DH_ + d];
    }
    __syncthreads();

    float qv[DH_];
    #pragma unroll
    for (int d = 0; d < DH_; ++d)
        qv[d] = Qb[(size_t)(b * Q_ + q) * qStride + h * DH_ + d];

    float s[NK];
    float mx = -1e30f;
    for (int k = 0; k < NK; ++k) {
        float dot = 0.f;
        #pragma unroll
        for (int d = 0; d < DH_; ++d) dot += qv[d] * KV[k][d];
        dot *= 0.125f;                 // 1/sqrt(64)
        s[k] = dot;
        mx = fmaxf(mx, dot);
    }
    float sum = 0.f;
    for (int k = 0; k < NK; ++k) { float e = __expf(s[k] - mx); s[k] = e; sum += e; }
    const float inv = 1.f / sum;

    __syncthreads();
    for (int i = threadIdx.x; i < NK * DH_; i += 64) {
        int r = i / DH_, d = i % DH_;
        KV[r][d] = Vb[(size_t)(b * NK + r) * kvStride + h * DH_ + d];
    }
    __syncthreads();

    for (int d = 0; d < DH_; ++d) {
        float acc = 0.f;
        for (int k = 0; k < NK; ++k) acc += s[k] * KV[k][d];
        out[(size_t)(b * Q_ + q) * C_ + h * DH_ + d] = acc * inv;
    }
}

// ---------------------------------------------------------------------------
// out = LayerNorm(res + delta) * g + b   (one row of C=384 per block)
// ---------------------------------------------------------------------------
__global__ __launch_bounds__(128) void add_ln_kernel(
    const float* __restrict__ res, const float* __restrict__ delta,
    const float* __restrict__ g, const float* __restrict__ bta,
    float* __restrict__ out)
{
    __shared__ float row[C_];
    __shared__ float red[128];
    const int r = blockIdx.x;
    const int t = threadIdx.x;

    float lsum = 0.f;
    for (int i = t; i < C_; i += 128) {
        float v = res[(size_t)r * C_ + i] + delta[(size_t)r * C_ + i];
        row[i] = v; lsum += v;
    }
    red[t] = lsum; __syncthreads();
    for (int o = 64; o > 0; o >>= 1) { if (t < o) red[t] += red[t + o]; __syncthreads(); }
    const float mu = red[0] / C_;
    __syncthreads();

    float lvar = 0.f;
    for (int i = t; i < C_; i += 128) { float dd = row[i] - mu; lvar += dd * dd; }
    red[t] = lvar; __syncthreads();
    for (int o = 64; o > 0; o >>= 1) { if (t < o) red[t] += red[t + o]; __syncthreads(); }
    const float rstd = rsqrtf(red[0] / C_ + 1e-5f);

    for (int i = t; i < C_; i += 128)
        out[(size_t)r * C_ + i] = (row[i] - mu) * rstd * g[i] + bta[i];
}

// ---------------------------------------------------------------------------
// In-place softmax over the L*P=20 sampling weights of each (b,q,h).
// ---------------------------------------------------------------------------
__global__ void softmax20_kernel(float* __restrict__ aw)
{
    int t = blockIdx.x * blockDim.x + threadIdx.x;
    if (t >= B_ * Q_ * H_) return;
    int bq = t / H_, h = t % H_;
    float* p = aw + (size_t)bq * (H_ * L_ * P_) + h * (L_ * P_);
    float mx = -1e30f;
    #pragma unroll
    for (int i = 0; i < L_ * P_; ++i) mx = fmaxf(mx, p[i]);
    float e[L_ * P_], sum = 0.f;
    #pragma unroll
    for (int i = 0; i < L_ * P_; ++i) { e[i] = __expf(p[i] - mx); sum += e[i]; }
    const float inv = 1.f / sum;
    #pragma unroll
    for (int i = 0; i < L_ * P_; ++i) p[i] = e[i] * inv;
}

// ---------------------------------------------------------------------------
// Fused gather + value-projection + weighted-combine deformable attention.
// Block = (b,q,h), 256 threads (8 waves). Instead of projecting all B*S rows
// of src (78.8 GFLOP, 820MB traffic) we project only the 80 gathered bilinear
// corner rows per block through the 64-wide head slice of ms_val_w with WMMA
// (M=80, N=64, K=384): 13x fewer FLOPs, ~4x less HBM traffic, no 410MB
// intermediate. OOB corners: clamped gather index + zero combined weight.
// ---------------------------------------------------------------------------
#define NCORN 80   // L*P samples * 4 corners

__global__ __launch_bounds__(256) void ms_fused_kernel(
    const float* __restrict__ src,     // (B,S,C)
    const float* __restrict__ Wv,      // (C,C) row-major [n][k]
    const float* __restrict__ bv,      // (C)
    const float* __restrict__ offs,    // (B*Q, H*L*P*2)
    const float* __restrict__ aw,      // (B*Q, H*L*P), softmaxed
    const float* __restrict__ refp,    // (B,Q,2)
    const float* __restrict__ vratio,  // (B,L,2)
    const int* __restrict__ spatial, const int* __restrict__ lsi,
    float* __restrict__ samp)          // (B*Q, C)
{
    __shared__ int    sIdx[NCORN];          // gathered global row (b*S + s)
    __shared__ float  cW[NCORN];            // aw * bilinear corner weight
    __shared__ __bf16 As[NCORN][BK + 2];
    __shared__ __bf16 Bs[DH_][BK + 2];
    __shared__ float  Ps[NCORN][DH_ + 1];   // projected corners

    const int bqh = blockIdx.x;
    const int h   = bqh % H_;
    const int bq  = bqh / H_;
    const int b   = bq / Q_;

    const int tid   = threadIdx.x;
    const int wave  = tid >> 5;
    const int lane  = tid & 31;
    const int lhalf = lane >> 4;
    const int l16   = lane & 15;

    // ---- sample setup: 20 (l,p) samples -> 80 corners ----
    if (tid < L_ * P_) {
        const int l = tid / P_, p = tid % P_;
        const int hh = spatial[2 * l + 0];
        const int ww = spatial[2 * l + 1];
        const int s0 = lsi[l];
        const float rx = refp[(size_t)bq * 2 + 0] * vratio[(size_t)(b * L_ + l) * 2 + 0];
        const float ry = refp[(size_t)bq * 2 + 1] * vratio[(size_t)(b * L_ + l) * 2 + 1];
        const int oidx = bq * (H_ * L_ * P_ * 2) + ((h * L_ + l) * P_ + p) * 2;
        const float lx = rx + offs[oidx + 0] / (float)ww;
        const float ly = ry + offs[oidx + 1] / (float)hh;
        const float a  = aw[bq * (H_ * L_ * P_) + (h * L_ + l) * P_ + p];
        const float x = lx * (float)ww - 0.5f;
        const float y = ly * (float)hh - 0.5f;
        const float x0f = floorf(x), y0f = floorf(y);
        const float wx = x - x0f, wy = y - y0f;
        const int x0 = (int)x0f, y0 = (int)y0f;
        #pragma unroll
        for (int c = 0; c < 4; ++c) {
            const int xi = x0 + (c & 1);
            const int yi = y0 + (c >> 1);
            const float wc = ((c & 1) ? wx : 1.f - wx) * ((c >> 1) ? wy : 1.f - wy);
            const bool ok = (xi >= 0) && (xi < ww) && (yi >= 0) && (yi < hh);
            sIdx[tid * 4 + c] = b * S_ + (ok ? (s0 + yi * ww + xi) : 0);
            cW[tid * 4 + c]   = ok ? a * wc : 0.f;
        }
    }
    __syncthreads();

    // ---- WMMA projection of the 80 gathered rows (M=80, N=64, K=384) ----
    v8f acc0 = {}, acc1 = {}, acc2 = {}, acc3 = {};
    for (int k0 = 0; k0 < C_; k0 += BK) {
        for (int i = tid; i < NCORN * BK; i += 256) {
            int r = i >> 5, k = i & 31;
            As[r][k] = f2bf(src[(size_t)sIdx[r] * C_ + k0 + k]);
        }
        for (int i = tid; i < DH_ * BK; i += 256) {
            int n = i >> 5, k = i & 31;
            Bs[n][k] = f2bf(Wv[(size_t)(h * DH_ + n) * C_ + k0 + k]);
        }
        __syncthreads();

        if (wave < 5) {                       // 5 M-tiles of 16 rows
            union { v16bf v; __bf16 e[16]; } af;
            #pragma unroll
            for (int j = 0; j < 16; ++j) {
                int kk = ((j >= 8) ? 16 : 0) + (lhalf ? 8 : 0) + (j & 7);
                af.e[j] = As[wave * 16 + l16][kk];
            }
            #pragma unroll
            for (int ns = 0; ns < 4; ++ns) {
                union { v16bf v; __bf16 e[16]; } bfr;
                #pragma unroll
                for (int j = 0; j < 16; ++j) {
                    int kk = ((j >= 8) ? 16 : 0) + (lhalf ? 8 : 0) + (j & 7);
                    bfr.e[j] = Bs[ns * 16 + l16][kk];
                }
                v8f* accp = (ns == 0) ? &acc0 : (ns == 1) ? &acc1
                           : (ns == 2) ? &acc2 : &acc3;
                *accp = __builtin_amdgcn_wmma_f32_16x16x32_bf16(
                    false, af.v, false, bfr.v, (short)0, *accp, false, false);
            }
        }
        __syncthreads();
    }

    // ---- spill projected corners (+bias) to LDS ----
    if (wave < 5) {
        union { v8f v; float e[8]; } a0, a1, a2, a3;
        a0.v = acc0; a1.v = acc1; a2.v = acc2; a3.v = acc3;
        #pragma unroll
        for (int ns = 0; ns < 4; ++ns) {
            const float* ep = (ns == 0) ? a0.e : (ns == 1) ? a1.e
                             : (ns == 2) ? a2.e : a3.e;
            const int col = ns * 16 + l16;
            const float bvv = bv[h * DH_ + col];
            #pragma unroll
            for (int r = 0; r < 8; ++r)
                Ps[wave * 16 + r + lhalf * 8][col] = ep[r] + bvv;
        }
    }
    __syncthreads();

    // ---- weighted combine over the 80 corners ----
    if (tid < DH_) {
        float acc = 0.f;
        #pragma unroll 4
        for (int r = 0; r < NCORN; ++r) acc += cW[r] * Ps[r][tid];
        samp[(size_t)bq * C_ + h * DH_ + tid] = acc;
    }
}

// ---------------------------------------------------------------------------
extern "C" void kernel_launch(void* const* d_in, const int* in_sizes, int n_in,
                              void* d_out, int out_size, void* d_ws, size_t ws_size,
                              hipStream_t stream)
{
    (void)in_sizes; (void)n_in; (void)out_size;

    const float* tgt       = (const float*)d_in[0];
    const float* refp      = (const float*)d_in[1];
    const float* src       = (const float*)d_in[2];
    const float* extra     = (const float*)d_in[3];
    const float* vrat      = (const float*)d_in[4];
    const int*   spat      = (const int*)d_in[5];
    const int*   lsi       = (const int*)d_in[6];
    const float* sa_in_w   = (const float*)d_in[7];
    const float* sa_out_w  = (const float*)d_in[8];
    const float* ea_in_w   = (const float*)d_in[9];
    const float* ea_out_w  = (const float*)d_in[10];
    const float* ms_off_w  = (const float*)d_in[11];
    const float* ms_attn_w = (const float*)d_in[12];
    const float* ms_val_w  = (const float*)d_in[13];
    const float* ms_out_w  = (const float*)d_in[14];
    const float* ffn_w1    = (const float*)d_in[15];
    const float* ffn_w2    = (const float*)d_in[16];
    const float* sa_in_b   = (const float*)d_in[17];
    const float* sa_out_b  = (const float*)d_in[18];
    const float* ea_in_b   = (const float*)d_in[19];
    const float* ea_out_b  = (const float*)d_in[20];
    const float* ms_attn_b = (const float*)d_in[21];
    const float* ms_val_b  = (const float*)d_in[22];
    const float* ms_out_b  = (const float*)d_in[23];
    const float* ffn_b1    = (const float*)d_in[24];
    const float* ffn_b2    = (const float*)d_in[25];
    const float* ln2_b     = (const float*)d_in[26];
    const float* lne_b     = (const float*)d_in[27];
    const float* ln1_b     = (const float*)d_in[28];
    const float* ln3_b     = (const float*)d_in[29];
    const float* ms_off_b  = (const float*)d_in[30];
    const float* ln2_g     = (const float*)d_in[31];
    const float* lne_g     = (const float*)d_in[32];
    const float* ln1_g     = (const float*)d_in[33];
    const float* ln3_g     = (const float*)d_in[34];

    float* ws = (float*)d_ws;
    const int BQ = B_ * Q_;
    size_t o = 0;
    float* qkv_sa = ws + o; o += (size_t)BQ * 3 * C_;
    float* attn_t = ws + o; o += (size_t)BQ * C_;
    float* proj_t = ws + o; o += (size_t)BQ * C_;
    float* tgt_a  = ws + o; o += (size_t)BQ * C_;
    float* q_ea   = ws + o; o += (size_t)BQ * C_;
    float* kv_ea  = ws + o; o += (size_t)B_ * EXTRA_ * 2 * C_;
    float* tgt_b  = ws + o; o += (size_t)BQ * C_;
    float* offs   = ws + o; o += (size_t)BQ * H_ * L_ * P_ * 2;
    float* aw     = ws + o; o += (size_t)BQ * H_ * L_ * P_;
    float* samp   = ws + o; o += (size_t)BQ * C_;
    float* tgt_c  = ws + o; o += (size_t)BQ * C_;
    float* ffn_h  = ws + o; o += (size_t)BQ * DFF_;
    if (ws_size < o * sizeof(float)) return;   // ~7MB total now (was ~417MB)

    auto gemm = [&](const float* X, const float* W, const float* bias, float* Y,
                    int M, int N, int K, bool relu) {
        dim3 grid((N + BN - 1) / BN, (M + BM - 1) / BM);
        if (relu) gemm_bias_wmma<true ><<<grid, 256, 0, stream>>>(X, W, bias, Y, M, N, K);
        else      gemm_bias_wmma<false><<<grid, 256, 0, stream>>>(X, W, bias, Y, M, N, K);
    };

    // ---- self attention ----
    gemm(tgt, sa_in_w, sa_in_b, qkv_sa, BQ, 3 * C_, C_, false);
    attn_kernel<Q_><<<B_ * H_, 64, 0, stream>>>(qkv_sa, 3 * C_,
                                                qkv_sa + C_, qkv_sa + 2 * C_, 3 * C_, attn_t);
    gemm(attn_t, sa_out_w, sa_out_b, proj_t, BQ, C_, C_, false);
    add_ln_kernel<<<BQ, 128, 0, stream>>>(tgt, proj_t, ln2_g, ln2_b, tgt_a);

    // ---- cross attention with extra memory ----
    gemm(tgt_a, ea_in_w, ea_in_b, q_ea, BQ, C_, C_, false);
    gemm(extra, ea_in_w + (size_t)C_ * C_, ea_in_b + C_, kv_ea,
         B_ * EXTRA_, 2 * C_, C_, false);
    attn_kernel<EXTRA_><<<B_ * H_, 64, 0, stream>>>(q_ea, C_,
                                                    kv_ea, kv_ea + C_, 2 * C_, attn_t);
    gemm(attn_t, ea_out_w, ea_out_b, proj_t, BQ, C_, C_, false);
    add_ln_kernel<<<BQ, 128, 0, stream>>>(tgt_a, proj_t, lne_g, lne_b, tgt_b);

    // ---- multi-scale deformable attention (fused gather+project+combine) ----
    gemm(tgt_b, ms_off_w, ms_off_b, offs, BQ, H_ * L_ * P_ * 2, C_, false);
    gemm(tgt_b, ms_attn_w, ms_attn_b, aw, BQ, H_ * L_ * P_, C_, false);
    softmax20_kernel<<<(B_ * Q_ * H_ + 255) / 256, 256, 0, stream>>>(aw);
    ms_fused_kernel<<<B_ * Q_ * H_, 256, 0, stream>>>(src, ms_val_w, ms_val_b,
                                                      offs, aw, refp, vrat,
                                                      spat, lsi, samp);
    gemm(samp, ms_out_w, ms_out_b, proj_t, BQ, C_, C_, false);
    add_ln_kernel<<<BQ, 128, 0, stream>>>(tgt_b, proj_t, ln1_g, ln1_b, tgt_c);

    // ---- FFN ----
    gemm(tgt_c, ffn_w1, ffn_b1, ffn_h, BQ, DFF_, C_, true);
    gemm(ffn_h, ffn_w2, ffn_b2, proj_t, BQ, C_, DFF_, false);
    add_ln_kernel<<<BQ, 128, 0, stream>>>(tgt_c, proj_t, ln3_g, ln3_b, (float*)d_out);
}
